// Attention_46110768890377
// MI455X (gfx1250) — compile-verified, matrix operand order checked by reading, not run on hardware
//
#include <hip/hip_runtime.h>
#include <hip/hip_bf16.h>

// ---------------------------------------------------------------------------
// Problem constants: B=2, Nq=512, Nk=4096, Nk_new=1024, D=1024, H=16, DH=64,
// INNER=1024, NTOT=5120, SCALE=0.125
// ---------------------------------------------------------------------------

typedef __attribute__((ext_vector_type(16))) __bf16 v16bf;
typedef __attribute__((ext_vector_type(8)))  float  v8f;
typedef unsigned su4 __attribute__((ext_vector_type(4)));
typedef int      si8 __attribute__((ext_vector_type(8)));
typedef int      si4 __attribute__((ext_vector_type(4)));

__device__ __forceinline__ __bf16 f2bf(float f) {
  union { float f; unsigned u; } a; a.f = f;
  unsigned r = (a.u + 0x7FFFu + ((a.u >> 16) & 1u)) >> 16;
  union { unsigned short s; __bf16 b; } o; o.s = (unsigned short)r;
  return o.b;
}

__device__ __forceinline__ unsigned short f2bf_u16(float f) {
  union { float f; unsigned u; } a; a.f = f;
  return (unsigned short)((a.u + 0x7FFFu + ((a.u >> 16) & 1u)) >> 16);
}

// K-offset within a 16-bit A/B fragment: VGPR i packs bf16 pair (k, k+1),
// k = (i>=4 ? 16 : 0) + (i%4)*2 + (lane/16)*8   (per ISA 7.12.2)
__device__ __forceinline__ int kfrag(int i, int lk) {
  return ((i & 4) << 2) | ((i & 3) << 1) | (lk << 3);
}

// gfx1250 async copy: per-lane 16B global -> LDS, tracked by ASYNCcnt.
__device__ __forceinline__ void async_load_b128(unsigned lds_addr, const void* gaddr) {
  asm volatile("global_load_async_to_lds_b128 %0, %1, off"
               :: "v"(lds_addr), "v"((unsigned long long)(size_t)gaddr)
               : "memory");
}

// gfx1250 Tensor Data Mover: DMA a 2D tile (tileW x tileH elements, 2B each)
// of a row-major tensor (row stride = strideElems) into LDS at lds_addr.
// D# layout per CDNA5 ISA. Issued per-wave;
// completion tracked by TENSORcnt. This toolchain exposes the 6-arg builtin
// (g0, g1, g2, g3, extra-si8, cpol); the trailing si8 is zero for 2D tiles.
__device__ __forceinline__ void tdm_load_2d_bf16(
    unsigned lds_addr, const void* gaddr,
    unsigned tileW, unsigned tileH,
    unsigned strideElems, unsigned tensorW, unsigned tensorH) {
  unsigned long long ga = (unsigned long long)(size_t)gaddr;
  su4 g0;
  g0[0] = 1u;                                   // count=1 (valid), user mode
  g0[1] = lds_addr;                             // bits 63:32  lds_addr
  g0[2] = (unsigned)(ga & 0xFFFFFFFFu);         // bits 95:64  global_addr lo
  g0[3] = (unsigned)((ga >> 32) & 0x01FFFFFFu)  // bits 120:96 global_addr hi
        | (2u << 30);                           // bits 127:126 type=2 (image)
  si8 g1;
  g1[0] = (int)(1u << 16);                      // data_size=1 (2 bytes), mask=0
  g1[1] = (int)((tensorW & 0xFFFFu) << 16);     // tensor_dim0 lo16 @ bits 63:48
  g1[2] = (int)((tensorW >> 16) | ((tensorH & 0xFFFFu) << 16)); // dim0 hi | dim1 lo
  g1[3] = (int)((tensorH >> 16) | (tileW << 16));               // dim1 hi | tile_dim0
  g1[4] = (int)(tileH & 0xFFFFu);               // tile_dim1 (tile_dim2 = 0)
  g1[5] = (int)strideElems;                     // tensor_dim0_stride lo32
  g1[6] = 0;                                    // stride hi16 | dim1_stride lo16
  g1[7] = 0;
  si4 g2 = {0, 0, 0, 0};
  si4 g3 = {0, 0, 0, 0};
  si8 g4 = {0, 0, 0, 0, 0, 0, 0, 0};
  __builtin_amdgcn_tensor_load_to_lds(g0, g1, g2, g3, g4, 0);
}

// ---------------------------------------------------------------------------
// fp32 -> bf16 cast, 8 elements per thread
// ---------------------------------------------------------------------------
__global__ void cast_f32_bf16(const float* __restrict__ in,
                              __bf16* __restrict__ out, int n8) {
  int i = blockIdx.x * blockDim.x + threadIdx.x;
  if (i >= n8) return;
  const float4* src = (const float4*)(in + (size_t)i * 8);
  float4 a = src[0], b = src[1];
  union { uint4 v; unsigned short h[8]; } u;
  u.h[0] = f2bf_u16(a.x); u.h[1] = f2bf_u16(a.y);
  u.h[2] = f2bf_u16(a.z); u.h[3] = f2bf_u16(a.w);
  u.h[4] = f2bf_u16(b.x); u.h[5] = f2bf_u16(b.y);
  u.h[6] = f2bf_u16(b.z); u.h[7] = f2bf_u16(b.w);
  *(uint4*)(out + (size_t)i * 8) = u.v;
}

// ---------------------------------------------------------------------------
// bf16 WMMA GEMM: C[M,N] = A[M,K] * B[K,N] (row-major). Block tile 128x64,
// BK=64, 256 threads = 8 waves, wave tile 32x32 (2x2 WMMA, 8 wmma/k-stage).
// A tile staged by the Tensor Data Mover (one descriptor per stage, wave 0);
// B tile staged transposed by all threads.
// Epilogue modes (rowsPerBatch = 1<<rpbShift):
//   1: q proj   -> dst0 = bf16 q_head[B,H,Nq,64], value * 0.125
//   2: kv proj  -> dst0/dst1 = K/V head[B,H,5120,64], position = posoff + pos
//   3: out proj -> dst0 = float C + bias[col]
// ---------------------------------------------------------------------------
__global__ __launch_bounds__(256) void gemm_bf16_wmma(
    const __bf16* __restrict__ A, const __bf16* __restrict__ Bm,
    void* __restrict__ dst0, void* __restrict__ dst1,
    const float* __restrict__ bias,
    int M, int N, int K, int mode, int rpbShift, int posoff) {
  __shared__ __bf16 sA[128 * 64];   // [row][k]
  __shared__ __bf16 sB[64 * 64];    // transposed: [n][k]

  const int tid = threadIdx.x;
  const int w  = tid >> 5;
  const int L  = tid & 31;
  const int lm = L & 15, ln = L & 15, lk = L >> 4;
  const int m0w = (w >> 1) * 32;
  const int n0w = (w & 1) * 32;
  const int rowBase = blockIdx.y * 128;
  const int colBase = blockIdx.x * 64;
  const unsigned sA_lds = (unsigned)(size_t)(&sA[0]);

  v8f acc[2][2];
  acc[0][0] = (v8f){}; acc[0][1] = (v8f){};
  acc[1][0] = (v8f){}; acc[1][1] = (v8f){};

  for (int k0 = 0; k0 < K; k0 += 64) {
    __syncthreads();
    // Stage A tile 128x64 via TDM (single descriptor, issued by wave 0)
    if (w == 0) {
      tdm_load_2d_bf16(sA_lds, A + (size_t)rowBase * K + k0,
                       /*tileW=*/64, /*tileH=*/128,
                       /*strideElems=*/(unsigned)K,
                       /*tensorW=*/(unsigned)K, /*tensorH=*/(unsigned)M);
    }
    // Stage B tile 64x64, transposed into sB[n][k] (512 uint4, 2 per thread)
    #pragma unroll
    for (int r = 0; r < 2; ++r) {
      int idx = tid + r * 256;
      int kk = idx >> 3, ch = idx & 7;
      uint4 d = *(const uint4*)(Bm + (size_t)(k0 + kk) * N + colBase + ch * 8);
      union { uint4 v; unsigned short h[8]; } u; u.v = d;
      #pragma unroll
      for (int j = 0; j < 8; ++j)
        ((unsigned short*)sB)[(ch * 8 + j) * 64 + kk] = u.h[j];
    }
    if (w == 0) __builtin_amdgcn_s_wait_tensorcnt(0);
    __syncthreads();

    #pragma unroll
    for (int hh = 0; hh < 2; ++hh) {
      const int kk0 = hh * 32;
      v16bf a0, a1, b0, b1;
      #pragma unroll
      for (int i = 0; i < 8; ++i) {
        int k = kk0 + kfrag(i, lk);
        a0[2*i] = sA[(m0w + lm) * 64 + k];       a0[2*i+1] = sA[(m0w + lm) * 64 + k + 1];
        a1[2*i] = sA[(m0w + 16 + lm) * 64 + k];  a1[2*i+1] = sA[(m0w + 16 + lm) * 64 + k + 1];
        b0[2*i] = sB[(n0w + ln) * 64 + k];       b0[2*i+1] = sB[(n0w + ln) * 64 + k + 1];
        b1[2*i] = sB[(n0w + 16 + ln) * 64 + k];  b1[2*i+1] = sB[(n0w + 16 + ln) * 64 + k + 1];
      }
      acc[0][0] = __builtin_amdgcn_wmma_f32_16x16x32_bf16(false, a0, false, b0, (short)0, acc[0][0], false, false);
      acc[0][1] = __builtin_amdgcn_wmma_f32_16x16x32_bf16(false, a0, false, b1, (short)0, acc[0][1], false, false);
      acc[1][0] = __builtin_amdgcn_wmma_f32_16x16x32_bf16(false, a1, false, b0, (short)0, acc[1][0], false, false);
      acc[1][1] = __builtin_amdgcn_wmma_f32_16x16x32_bf16(false, a1, false, b1, (short)0, acc[1][1], false, false);
    }
  }

  // Epilogue (C layout: lane n = L%16, vgpr i holds row m = i + 8*(L/16))
  const int rpbMask = (1 << rpbShift) - 1;
  for (int si = 0; si < 2; ++si) {
    for (int sj = 0; sj < 2; ++sj) {
      int r0 = rowBase + m0w + si * 16;
      int c0 = colBase + n0w + sj * 16;
      #pragma unroll
      for (int i = 0; i < 8; ++i) {
        int grow = r0 + i + 8 * lk;
        int gcol = c0 + ln;
        float v = acc[si][sj][i];
        if (mode == 1) {
          int b = grow >> rpbShift, pos = grow & rpbMask;
          int h = gcol >> 6, dh = gcol & 63;
          size_t idx = ((((size_t)(b * 16 + h) << rpbShift) + pos) << 6) + dh;
          ((__bf16*)dst0)[idx] = f2bf(v * 0.125f);
        } else if (mode == 2) {
          int b = grow >> rpbShift, pos = grow & rpbMask;
          int c = gcol & 1023;
          int h = c >> 6, dh = c & 63;
          __bf16* dst = (gcol < 1024) ? (__bf16*)dst0 : (__bf16*)dst1;
          dst[(((size_t)(b * 16 + h) * 5120 + posoff + pos) << 6) + dh] = f2bf(v);
        } else {
          ((float*)dst0)[(size_t)grow * N + gcol] = v + bias[gcol];
        }
      }
    }
  }
}

// ---------------------------------------------------------------------------
// Flash attention over 5120 keys. Grid: (Nq/128, B*H). 256 threads = 8 waves,
// each wave owns 16 queries, key tiles of 32, online softmax.
// K tile staged via global_load_async_to_lds_b128 (ASYNCcnt path).
// ---------------------------------------------------------------------------
__global__ __launch_bounds__(256) void flash_attn_wmma(
    const __bf16* __restrict__ Q,   // [B,H,512,64]
    const __bf16* __restrict__ Kb,  // [B,H,5120,64]
    const __bf16* __restrict__ Vb,  // [B,H,5120,64]
    __bf16* __restrict__ O) {       // [B,512,1024]
  __shared__ __bf16 sK[32 * 64];       // [key][d]
  __shared__ __bf16 sV[64 * 32];       // transposed: [d][key]
  __shared__ __bf16 sP[8 * 16 * 32];   // per-wave P tile [m][key]

  const int tid = threadIdx.x;
  const int w  = tid >> 5;
  const int L  = tid & 31;
  const int lm = L & 15, ln = L & 15, lk = L >> 4;
  const int bh = blockIdx.y;
  const int q0 = blockIdx.x * 128 + w * 16;

  v16bf qa0, qa1;
  {
    const __bf16* qp = Q + ((size_t)bh * 512 + q0 + lm) * 64;
    #pragma unroll
    for (int i = 0; i < 8; ++i) {
      int k = kfrag(i, lk);
      qa0[2*i] = qp[k];       qa0[2*i+1] = qp[k + 1];
      qa1[2*i] = qp[32 + k];  qa1[2*i+1] = qp[32 + k + 1];
    }
  }

  float rmax[8], rsum[8];
  v8f o0 = (v8f){}, o1 = (v8f){}, o2 = (v8f){}, o3 = (v8f){};
  #pragma unroll
  for (int i = 0; i < 8; ++i) { rmax[i] = -1e30f; rsum[i] = 0.f; }

  __bf16* pw = sP + w * 512;
  const size_t kvbase = (size_t)bh * 5120 * 64;
  const int key = tid >> 3, ch = tid & 7;
  const unsigned sk_lds = (unsigned)(size_t)(sK + key * 64 + ch * 8);

  for (int kt = 0; kt < 160; ++kt) {
    __syncthreads();
    {
      size_t g = kvbase + (size_t)(kt * 32 + key) * 64 + ch * 8;
      // K tile: one async b128 per lane = whole 4KB tile, no VGPR round-trip
      async_load_b128(sk_lds, Kb + g);
      // V tile: load + transpose-scatter (needs [d][key] layout for B frags)
      uint4 vd = *(const uint4*)(Vb + g);
      union { uint4 v; unsigned short h[8]; } u; u.v = vd;
      #pragma unroll
      for (int j = 0; j < 8; ++j)
        ((unsigned short*)sV)[(ch * 8 + j) * 32 + key] = u.h[j];
      if (kt + 1 < 160) {
        __builtin_prefetch(Kb + g + 2048, 0, 0);
        __builtin_prefetch(Vb + g + 2048, 0, 0);
      }
    }
    asm volatile("s_wait_asynccnt 0" ::: "memory");
    __syncthreads();

    // S = Q * K^T : B-frags from K rows (lane = key, pairs along d)
    v16bf bkl0, bkh0, bkl1, bkh1;
    #pragma unroll
    for (int i = 0; i < 8; ++i) {
      int k = kfrag(i, lk);
      bkl0[2*i] = sK[ln * 64 + k];             bkl0[2*i+1] = sK[ln * 64 + k + 1];
      bkh0[2*i] = sK[ln * 64 + 32 + k];        bkh0[2*i+1] = sK[ln * 64 + 32 + k + 1];
      bkl1[2*i] = sK[(16 + ln) * 64 + k];      bkl1[2*i+1] = sK[(16 + ln) * 64 + k + 1];
      bkh1[2*i] = sK[(16 + ln) * 64 + 32 + k]; bkh1[2*i+1] = sK[(16 + ln) * 64 + 32 + k + 1];
    }
    v8f s0 = (v8f){}, s1 = (v8f){};
    s0 = __builtin_amdgcn_wmma_f32_16x16x32_bf16(false, qa0, false, bkl0, (short)0, s0, false, false);
    s0 = __builtin_amdgcn_wmma_f32_16x16x32_bf16(false, qa1, false, bkh0, (short)0, s0, false, false);
    s1 = __builtin_amdgcn_wmma_f32_16x16x32_bf16(false, qa0, false, bkl1, (short)0, s1, false, false);
    s1 = __builtin_amdgcn_wmma_f32_16x16x32_bf16(false, qa1, false, bkh1, (short)0, s1, false, false);

    // Online softmax in C-layout; 16-lane row reductions (halves independent)
    #pragma unroll
    for (int i = 0; i < 8; ++i) {
      float mx = fmaxf(s0[i], s1[i]);
      mx = fmaxf(mx, __shfl_xor(mx, 1));
      mx = fmaxf(mx, __shfl_xor(mx, 2));
      mx = fmaxf(mx, __shfl_xor(mx, 4));
      mx = fmaxf(mx, __shfl_xor(mx, 8));
      float nm = fmaxf(rmax[i], mx);
      float corr = __expf(rmax[i] - nm);
      rmax[i] = nm;
      float p0 = __expf(s0[i] - nm);
      float p1 = __expf(s1[i] - nm);
      float rs = p0 + p1;
      rs += __shfl_xor(rs, 1);
      rs += __shfl_xor(rs, 2);
      rs += __shfl_xor(rs, 4);
      rs += __shfl_xor(rs, 8);
      rsum[i] = rsum[i] * corr + rs;
      o0[i] *= corr; o1[i] *= corr; o2[i] *= corr; o3[i] *= corr;
      int m = i + 8 * lk;
      pw[m * 32 + ln]      = f2bf(p0);
      pw[m * 32 + 16 + ln] = f2bf(p1);
    }
    asm volatile("s_wait_dscnt 0" ::: "memory");  // wave-private P re-read below

    // P (16x32) as one A-fragment; V^T B-frags (lane = d col, pairs along key)
    v16bf pa, bv0, bv1, bv2, bv3;
    #pragma unroll
    for (int i = 0; i < 8; ++i) {
      int k = kfrag(i, lk);
      pa[2*i]  = pw[lm * 32 + k];            pa[2*i+1]  = pw[lm * 32 + k + 1];
      bv0[2*i] = sV[(ln)      * 32 + k];     bv0[2*i+1] = sV[(ln)      * 32 + k + 1];
      bv1[2*i] = sV[(16 + ln) * 32 + k];     bv1[2*i+1] = sV[(16 + ln) * 32 + k + 1];
      bv2[2*i] = sV[(32 + ln) * 32 + k];     bv2[2*i+1] = sV[(32 + ln) * 32 + k + 1];
      bv3[2*i] = sV[(48 + ln) * 32 + k];     bv3[2*i+1] = sV[(48 + ln) * 32 + k + 1];
    }
    o0 = __builtin_amdgcn_wmma_f32_16x16x32_bf16(false, pa, false, bv0, (short)0, o0, false, false);
    o1 = __builtin_amdgcn_wmma_f32_16x16x32_bf16(false, pa, false, bv1, (short)0, o1, false, false);
    o2 = __builtin_amdgcn_wmma_f32_16x16x32_bf16(false, pa, false, bv2, (short)0, o2, false, false);
    o3 = __builtin_amdgcn_wmma_f32_16x16x32_bf16(false, pa, false, bv3, (short)0, o3, false, false);
  }

  // Normalize, store O as [b, pos, h*64 + d] (row-major for the final GEMM)
  const int b = bh >> 4, h = bh & 15;
  #pragma unroll
  for (int i = 0; i < 8; ++i) {
    float inv = 1.0f / rsum[i];
    int m = i + 8 * lk;
    size_t orow = ((size_t)b * 512 + q0 + m) * 1024 + (size_t)h * 64;
    O[orow + 0  + ln] = f2bf(o0[i] * inv);
    O[orow + 16 + ln] = f2bf(o1[i] * inv);
    O[orow + 32 + ln] = f2bf(o2[i] * inv);
    O[orow + 48 + ln] = f2bf(o3[i] * inv);
  }
}

// ---------------------------------------------------------------------------
// Host-side orchestration
// ---------------------------------------------------------------------------
extern "C" void kernel_launch(void* const* d_in, const int* in_sizes, int n_in,
                              void* d_out, int out_size, void* d_ws, size_t ws_size,
                              hipStream_t stream) {
  const float* x    = (const float*)d_in[0];
  const float* ctx  = (const float*)d_in[1];
  const float* ctxn = (const float*)d_in[2];
  const float* Wq   = (const float*)d_in[3];
  const float* Wkv  = (const float*)d_in[4];
  const float* Wout = (const float*)d_in[5];
  const float* bout = (const float*)d_in[6];
  float* out = (float*)d_out;

  char* ws = (char*)d_ws;
  size_t off = 0;
  auto alloc = [&](size_t bytes) -> void* {
    void* p = ws + off;
    off += (bytes + 255) & ~(size_t)255;
    return p;
  };

  __bf16* xb    = (__bf16*)alloc((size_t)1024 * 1024 * 2);   // [B*Nq, D]
  __bf16* ctxb  = (__bf16*)alloc((size_t)8192 * 1024 * 2);   // [B*Nk, D]
  __bf16* ctxnb = (__bf16*)alloc((size_t)2048 * 1024 * 2);   // [B*Nk_new, D]
  __bf16* Wqb   = (__bf16*)alloc((size_t)1024 * 1024 * 2);
  __bf16* Wkvb  = (__bf16*)alloc((size_t)1024 * 2048 * 2);
  __bf16* Woutb = (__bf16*)alloc((size_t)1024 * 1024 * 2);
  __bf16* qh    = (__bf16*)alloc((size_t)2 * 16 * 512 * 64 * 2);   // [B,H,Nq,64]
  __bf16* Kh    = (__bf16*)alloc((size_t)2 * 16 * 5120 * 64 * 2);  // [B,H,5120,64]
  __bf16* Vh    = (__bf16*)alloc((size_t)2 * 16 * 5120 * 64 * 2);
  __bf16* Obf   = (__bf16*)alloc((size_t)1024 * 1024 * 2);         // [B*Nq, inner]

  auto cast = [&](const float* s, __bf16* d, int n) {
    int n8 = n / 8;
    cast_f32_bf16<<<(n8 + 255) / 256, 256, 0, stream>>>(s, d, n8);
  };
  cast(x,    xb,    1024 * 1024);
  cast(ctx,  ctxb,  8192 * 1024);
  cast(ctxn, ctxnb, 2048 * 1024);
  cast(Wq,   Wqb,   1024 * 1024);
  cast(Wkv,  Wkvb,  1024 * 2048);
  cast(Wout, Woutb, 1024 * 1024);

  dim3 blk(256);
  // q = x @ Wq (scaled 0.125, head-major scatter); rowsPerBatch = 512 = 1<<9
  gemm_bf16_wmma<<<dim3(1024 / 64, 1024 / 128), blk, 0, stream>>>(
      xb, Wqb, qh, nullptr, nullptr, 1024, 1024, 1024, 1, 9, 0);
  // kv = ctx @ Wkv -> positions [0,4096); rowsPerBatch = 4096 = 1<<12
  gemm_bf16_wmma<<<dim3(2048 / 64, 8192 / 128), blk, 0, stream>>>(
      ctxb, Wkvb, Kh, Vh, nullptr, 8192, 2048, 1024, 2, 12, 0);
  // kv2 = ctx_new @ Wkv -> positions [4096,5120); rowsPerBatch = 1024 = 1<<10
  gemm_bf16_wmma<<<dim3(2048 / 64, 2048 / 128), blk, 0, stream>>>(
      ctxnb, Wkvb, Kh, Vh, nullptr, 2048, 2048, 1024, 2, 10, 4096);
  // Flash attention over all 5120 keys
  flash_attn_wmma<<<dim3(512 / 128, 2 * 16), blk, 0, stream>>>(qh, Kh, Vh, Obf);
  // out = O @ Wout + bout (fp32)
  gemm_bf16_wmma<<<dim3(1024 / 64, 1024 / 128), blk, 0, stream>>>(
      Obf, Woutb, out, nullptr, bout, 1024, 1024, 1024, 3, 9, 0);
}
